// EquiformerV2_MatPES_71923522339384
// MI455X (gfx1250) — compile-verified, matrix operand order checked by reading, not run on hardware
//
#include <hip/hip_runtime.h>
#include <hip/hip_bf16.h>

typedef __attribute__((ext_vector_type(16))) __bf16 v16bf;
typedef __attribute__((ext_vector_type(8)))  float  v8f;

#define KPAD   864
#define KT1    27
#define LDS_W  14080            // ushorts per wave: 512B header + 27*1KB frags
#define WAVES  2

__device__ __forceinline__ unsigned short f2bf(float f) {
  unsigned int u = __float_as_uint(f);
  u += 0x7FFFu + ((u >> 16) & 1u);          // round-to-nearest-even bf16
  return (unsigned short)(u >> 16);
}

// 16-bit A-matrix 16x32 fragment layout (ISA 7.12.2):
// lanes 0-15: M=lane, elems 0-7 -> K'=0-7, elems 8-15 -> K'=16-23
// lanes 16-31: M=lane-16, elems 0-7 -> K'=8-15, elems 8-15 -> K'=24-31
__device__ __forceinline__ int afrag_addr(int m, int k) {
  int kt = k >> 5, kp = k & 31;
  int ln = m + (((kp & 15) >= 8) << 4);
  int el = (kp & 7) + ((kp >= 16) << 3);
  return kt * 512 + ln * 16 + el;            // ushort index
}

// ---------- pack a KxN fp32 row-major weight into bf16 B-fragments ----------
__global__ void pack_b_kernel(const float* __restrict__ W, unsigned short* __restrict__ dst,
                              int Ksrc, int Kpad, int Ncols) {
  int idx = blockIdx.x * blockDim.x + threadIdx.x;
  int total = Kpad * Ncols;
  if (idx >= total) return;
  int k = idx / Ncols, n = idx % Ncols;
  float v = (k < Ksrc) ? W[k * Ncols + n] : 0.f;
  int kt = k >> 5, kp = k & 31, nt = n >> 4;
  int ln = (n & 15) + (((kp & 15) >= 8) << 4);   // B mirrors A with N in lanes
  int el = (kp & 7) + ((kp >= 16) << 3);
  int NT = Ncols >> 4;
  dst[(kt * NT + nt) * 512 + ln * 16 + el] = f2bf(v);
}

// ---------- init output: l=0 slot gets sphere_emb, l=1 slots zero ----------
__global__ void init_out_kernel(float* __restrict__ out, const float* __restrict__ sph,
                                const int* __restrict__ atnum, int Nn) {
  int idx = blockIdx.x * blockDim.x + threadIdx.x;
  if (idx >= Nn * 512) return;
  int row = idx >> 9, rem = idx & 511;
  out[idx] = (rem < 128) ? sph[atnum[row] * 128 + rem] : 0.f;
}

// ---------- LN(+bias) + SiLU in WMMA C-layout, emit bf16 A-fragments ----------
__device__ __forceinline__ void ln_silu_store(v8f* acc, const float* bv, const float* gv,
                                              const float* bev, unsigned short* dstf, int lane) {
  int n = lane & 15, hs = lane >> 4;
#pragma unroll
  for (int r = 0; r < 8; ++r) {
    float s = 0.f, q = 0.f;
#pragma unroll
    for (int nt = 0; nt < 8; ++nt) {
      float x = acc[nt][r] + bv[nt];
      acc[nt][r] = x; s += x; q += x * x;
    }
#pragma unroll
    for (int msk = 1; msk <= 8; msk <<= 1) {     // butterfly within 16-lane half
      s += __shfl_xor(s, msk, 32);
      q += __shfl_xor(q, msk, 32);
    }
    float mu  = s * 0.0078125f;
    float inv = rsqrtf(fmaxf(q * 0.0078125f - mu * mu, 0.f) + 1e-5f);
    int M = r + (hs << 3);
#pragma unroll
    for (int nt = 0; nt < 8; ++nt) {
      float x = (acc[nt][r] - mu) * inv * gv[nt] + bev[nt];
      x = x / (1.f + __expf(-x));                // SiLU
      dstf[afrag_addr(M, 16 * nt + n)] = f2bf(x);
    }
  }
}

// =========================== fused edge kernel ===========================
__global__ __launch_bounds__(WAVES * 32)
void eqv2_edge_kernel(const float* __restrict__ evec,
                      const float* __restrict__ src_emb, const float* __restrict__ tgt_emb,
                      const float* __restrict__ b1, const float* __restrict__ g1, const float* __restrict__ be1,
                      const float* __restrict__ b2, const float* __restrict__ g2, const float* __restrict__ be2,
                      const float* __restrict__ b3,
                      const int* __restrict__ atnum, const int* __restrict__ eidx,
                      const unsigned short* __restrict__ w1p,
                      const unsigned short* __restrict__ w2p,
                      const unsigned short* __restrict__ w3p,
                      float* __restrict__ out, int E) {
  __shared__ __align__(64) unsigned short lds[WAVES * LDS_W];
  const int lane = threadIdx.x & 31;
  const int wave = threadIdx.x >> 5;
  unsigned short* base  = &lds[wave * LDS_W];
  float*          hf    = (float*)base;          // dist[16] d1x[16] d1y[16] d1z[16]
  int*            hi    = (int*)base + 64;       // zs[16] zt[16] dst[16]
  unsigned short* afrag = base + 256;
  const int cn = lane & 15, hs = lane >> 4;

  // per-lane channel constants (c = 16*j + cn)
  float b1v[8], g1v[8], be1v[8], b2v[8], g2v[8], be2v[8], b3v[16];
#pragma unroll
  for (int j = 0; j < 8; ++j) {
    b1v[j] = b1[16 * j + cn];  g1v[j] = g1[16 * j + cn];  be1v[j] = be1[16 * j + cn];
    b2v[j] = b2[16 * j + cn];  g2v[j] = g2[16 * j + cn];  be2v[j] = be2[16 * j + cn];
  }
#pragma unroll
  for (int j = 0; j < 16; ++j) b3v[j] = b3[16 * j + cn];

  const float DELTA  = 6.0f / 599.0f;
  const float IDELTA = 599.0f / 6.0f;
  const float COEF   = -0.5f / ((2.0f * DELTA) * (2.0f * DELTA));
  const int ntile = (E + 15) >> 4;

  for (int tile = blockIdx.x * WAVES + wave; tile < ntile; tile += gridDim.x * WAVES) {
    // ---- phase 0: per-edge geometry (lanes 0..15, one edge each) ----
    if (lane < 16) {
      int e = tile * 16 + lane;
      float dist = 0.f, d1x = 0.f, d1y = 0.f, d1z = 0.f;
      int zs = 0, zt = 0, dn = -1;
      if (e < E) {
        float ex = evec[3 * e], ey = evec[3 * e + 1], ez = evec[3 * e + 2];
        float nrm = sqrtf(ex * ex + ey * ey + ez * ez);
        dist = nrm;
        float dc = fmaxf(nrm, 1e-8f);
        float x0 = ex / dc, x1 = ey / dc, x2 = ez / dc;
        float a0 = fabsf(x0), a1 = fabsf(x1), a2 = fabsf(x2);
        float r0 = 0.f, r1 = 0.f, r2 = 0.f;
        if (a0 <= a1 && a0 <= a2) r0 = 1.f; else if (a1 <= a2) r1 = 1.f; else r2 = 1.f;
        float z0 = x1 * r2 - x2 * r1, z1 = x2 * r0 - x0 * r2, z2 = x0 * r1 - x1 * r0;
        float znm = fmaxf(sqrtf(z0 * z0 + z1 * z1 + z2 * z2), 1e-8f);
        z0 /= znm; z1 /= znm; z2 /= znm;
        float y0 = x1 * z2 - x2 * z1, y1 = x2 * z0 - x0 * z2, y2 = x0 * z1 - x1 * z0;
        float ynm = fmaxf(sqrtf(y0 * y0 + y1 * y1 + y2 * y2), 1e-8f);
        y0 /= ynm; y1 /= ynm; y2 /= ynm;
        d1x = -y1; d1y = -y2; d1z = -y0;       // D1 row 1 = (-ny) P^T
        int se = eidx[e], de = eidx[E + e];
        zs = atnum[se]; zt = atnum[de]; dn = de;
      }
      hf[lane] = dist; hf[16 + lane] = d1x; hf[32 + lane] = d1y; hf[48 + lane] = d1z;
      hi[lane] = zs;   hi[16 + lane] = zt;   hi[32 + lane] = dn;
    }
    asm volatile("s_wait_dscnt 0" ::: "memory");

    // ---- phase 1: build A tile (16x864 bf16) in fragment order in LDS ----
    {
      // 1a: zero the whole fragment region with wide DS stores (27*512 ushorts)
      uint4 zv; zv.x = zv.y = zv.z = zv.w = 0u;
      uint4* zp = (uint4*)afrag;
      for (int i = lane; i < (KT1 * 512) / 8; i += 32) zp[i] = zv;

      const int m = cn;
      float distm = hf[m];
      int zsm = hi[m], ztm = hi[16 + m];

      // 1b: Gaussian window — only |k - dist/delta| <= 34 survives bf16
      int kc = (int)(distm * IDELTA);
      kc = min(max(kc, 0), 599);
      int ks, ke;
      if (hs == 0) { ks = max(kc - 34, 0); ke = kc; }
      else         { ks = kc + 1;          ke = min(kc + 34, 599); }
      for (int k = ks; k <= ke; ++k) {
        float d = distm - (float)k * DELTA;
        afrag[afrag_addr(m, k)] = f2bf(__expf(COEF * d * d));
      }

      // 1c: embeddings — contiguous float4 loads (hs=0 -> src, hs=1 -> tgt)
      const float4* etab = (const float4*)((hs == 0) ? (src_emb + zsm * 128)
                                                     : (tgt_emb + ztm * 128));
      const int kbase = 600 + 128 * hs;
#pragma unroll 8
      for (int j = 0; j < 32; ++j) {
        float4 v4 = etab[j];
        int k = kbase + 4 * j;
        afrag[afrag_addr(m, k    )] = f2bf(v4.x);
        afrag[afrag_addr(m, k + 1)] = f2bf(v4.y);
        afrag[afrag_addr(m, k + 2)] = f2bf(v4.z);
        afrag[afrag_addr(m, k + 3)] = f2bf(v4.w);
      }
    }
    asm volatile("s_wait_dscnt 0" ::: "memory");

    v8f acc[8];
    v16bf bb[2][8];

    // ---- GEMM1: [16x864] @ [864x128], 27x8 bf16 WMMAs, double-buffered B ----
#pragma unroll
    for (int nt = 0; nt < 8; ++nt)
#pragma unroll
      for (int r = 0; r < 8; ++r) acc[nt][r] = 0.f;
#pragma unroll
    for (int nt = 0; nt < 8; ++nt)
      bb[0][nt] = *(const v16bf*)(w1p + nt * 512 + lane * 16);
#pragma unroll
    for (int kt = 0; kt < KT1; ++kt) {
      const int cur = kt & 1;
      if (kt + 1 < KT1) {
        const unsigned short* wn = w1p + (kt + 1) * 4096 + lane * 16;
#pragma unroll
        for (int nt = 0; nt < 8; ++nt)
          bb[cur ^ 1][nt] = *(const v16bf*)(wn + nt * 512);
      }
      v16bf a = *(const v16bf*)(afrag + kt * 512 + lane * 16);
#pragma unroll
      for (int nt = 0; nt < 8; ++nt)
        acc[nt] = __builtin_amdgcn_wmma_f32_16x16x32_bf16(false, a, false, bb[cur][nt],
                                                          (short)0, acc[nt], false, false);
    }
    ln_silu_store(acc, b1v, g1v, be1v, afrag, lane);          // h1 frags at offset 0
    asm volatile("s_wait_dscnt 0" ::: "memory");

    // ---- GEMM2: [16x128] @ [128x128], double-buffered B ----
#pragma unroll
    for (int nt = 0; nt < 8; ++nt)
#pragma unroll
      for (int r = 0; r < 8; ++r) acc[nt][r] = 0.f;
#pragma unroll
    for (int nt = 0; nt < 8; ++nt)
      bb[0][nt] = *(const v16bf*)(w2p + nt * 512 + lane * 16);
#pragma unroll
    for (int kt = 0; kt < 4; ++kt) {
      const int cur = kt & 1;
      if (kt + 1 < 4) {
        const unsigned short* wn = w2p + (kt + 1) * 4096 + lane * 16;
#pragma unroll
        for (int nt = 0; nt < 8; ++nt)
          bb[cur ^ 1][nt] = *(const v16bf*)(wn + nt * 512);
      }
      v16bf a = *(const v16bf*)(afrag + kt * 512 + lane * 16);
#pragma unroll
      for (int nt = 0; nt < 8; ++nt)
        acc[nt] = __builtin_amdgcn_wmma_f32_16x16x32_bf16(false, a, false, bb[cur][nt],
                                                          (short)0, acc[nt], false, false);
    }
    ln_silu_store(acc, b2v, g2v, be2v, afrag + 2048, lane);   // h2 frags at offset 2048
    asm volatile("s_wait_dscnt 0" ::: "memory");

    // ---- GEMM3: [16x128] @ [128x256] in two 128-col halves; scatter ----
#pragma unroll
    for (int half = 0; half < 2; ++half) {
#pragma unroll
      for (int nt = 0; nt < 8; ++nt)
#pragma unroll
        for (int r = 0; r < 8; ++r) acc[nt][r] = 0.f;
#pragma unroll
      for (int nt = 0; nt < 8; ++nt)
        bb[0][nt] = *(const v16bf*)(w3p + (half * 8 + nt) * 512 + lane * 16);
#pragma unroll
      for (int kt = 0; kt < 4; ++kt) {
        const int cur = kt & 1;
        if (kt + 1 < 4) {
          const unsigned short* wn = w3p + ((kt + 1) * 16 + half * 8) * 512 + lane * 16;
#pragma unroll
          for (int nt = 0; nt < 8; ++nt)
            bb[cur ^ 1][nt] = *(const v16bf*)(wn + nt * 512);
        }
        v16bf a = *(const v16bf*)(afrag + 2048 + kt * 512 + lane * 16);
#pragma unroll
        for (int nt = 0; nt < 8; ++nt)
          acc[nt] = __builtin_amdgcn_wmma_f32_16x16x32_bf16(false, a, false, bb[cur][nt],
                                                            (short)0, acc[nt], false, false);
      }
#pragma unroll
      for (int r = 0; r < 8; ++r) {
        int M = r + (hs << 3);
        int dn = hi[32 + M];
        if (dn < 0) continue;
        if (half == 0) {                                      // l=0 message
          float* op = out + (size_t)dn * 512;
#pragma unroll
          for (int nt = 0; nt < 8; ++nt) {
            float v = (acc[nt][r] + b3v[nt]) * (1.0f / 12.0f);
            atomicAdd(op + 16 * nt + cn, v);
          }
        } else {                                              // l=1: outer-prod with D1 row
          float dx = hf[16 + M], dy = hf[32 + M], dz = hf[48 + M];
          float* op = out + (size_t)dn * 512 + 128;
#pragma unroll
          for (int nt = 0; nt < 8; ++nt) {
            float v = (acc[nt][r] + b3v[8 + nt]) * (1.0f / 12.0f);
            atomicAdd(op +       16 * nt + cn, dx * v);
            atomicAdd(op + 128 + 16 * nt + cn, dy * v);
            atomicAdd(op + 256 + 16 * nt + cn, dz * v);
          }
        }
      }
    }
  }
}

extern "C" void kernel_launch(void* const* d_in, const int* in_sizes, int n_in,
                              void* d_out, int out_size, void* d_ws, size_t ws_size,
                              hipStream_t stream) {
  const float* evec    = (const float*)d_in[0];
  const float* sph     = (const float*)d_in[1];
  const float* src_emb = (const float*)d_in[2];
  const float* tgt_emb = (const float*)d_in[3];
  const float* W1      = (const float*)d_in[4];
  const float* b1      = (const float*)d_in[5];
  const float* g1      = (const float*)d_in[6];
  const float* be1     = (const float*)d_in[7];
  const float* W2      = (const float*)d_in[8];
  const float* b2      = (const float*)d_in[9];
  const float* g2      = (const float*)d_in[10];
  const float* be2     = (const float*)d_in[11];
  const float* W3      = (const float*)d_in[12];
  const float* b3      = (const float*)d_in[13];
  const int*   atnum   = (const int*)d_in[14];
  const int*   eidx    = (const int*)d_in[15];
  float*       out     = (float*)d_out;

  const int E  = in_sizes[0] / 3;
  const int Nn = in_sizes[14];

  unsigned short* w1p = (unsigned short*)d_ws;          // 864*128  bf16
  unsigned short* w2p = w1p + 864 * 128;                // 128*128  bf16
  unsigned short* w3p = w2p + 128 * 128;                // 128*256  bf16

  pack_b_kernel<<<(864 * 128 + 255) / 256, 256, 0, stream>>>(W1, w1p, 856, 864, 128);
  pack_b_kernel<<<(128 * 128 + 255) / 256, 256, 0, stream>>>(W2, w2p, 128, 128, 128);
  pack_b_kernel<<<(128 * 256 + 255) / 256, 256, 0, stream>>>(W3, w3p, 128, 128, 256);
  init_out_kernel<<<((size_t)Nn * 512 + 255) / 256, 256, 0, stream>>>(out, sph, atnum, Nn);

  int ntile  = (E + 15) / 16;
  int blocks = (ntile + WAVES - 1) / WAVES;
  if (blocks > 1536) blocks = 1536;
  if (blocks < 1)    blocks = 1;
  eqv2_edge_kernel<<<blocks, WAVES * 32, 0, stream>>>(
      evec, src_emb, tgt_emb, b1, g1, be1, b2, g2, be2, b3,
      atnum, eidx, w1p, w2p, w3p, out, E);
}
